// Head_25855703122234
// MI455X (gfx1250) — compile-verified, matrix operand order checked by reading, not run on hardware
//
#include <hip/hip_runtime.h>
#include <hip/hip_bf16.h>

// ---------------------------------------------------------------------------
// Single attention head, column-softmax (softmax over query axis, faithful to
// reference), causal mask. bf16 WMMA (16x16x32, f32 accum) for all 3 GEMMs.
// QKV GEMM: double-buffered LDS staging; X tile via TDM tensor_load_to_lds
// (TENSORcnt), W tile via per-lane global_load_async_to_lds_b128 (ASYNCcnt),
// fragments read from LDS via addrspace(3) loads (ds_load_b64).
// ---------------------------------------------------------------------------

typedef __bf16 bf16;
typedef __attribute__((ext_vector_type(16))) __bf16    v16bf;
typedef __attribute__((ext_vector_type(8)))  float     v8f;
typedef __attribute__((ext_vector_type(8)))  unsigned  v8u;
typedef __attribute__((ext_vector_type(4)))  unsigned  v4u;
typedef __attribute__((ext_vector_type(8)))  int       v8i;
typedef __attribute__((ext_vector_type(4)))  int       v4i;
typedef __attribute__((address_space(3))) const unsigned ldsCU;

#define Bn 16
#define Tn 2048
#define Cn 1024
#define Hn 64
#define CHUNK 256
#define NCHUNK (Tn / CHUNK)   // 8

// LDS tiling for the QKV GEMM (K-chunk = 32 bf16)
#define XR    40              // bf16 per LDS row: 32 data + 8 pad (80B, 16B-aligned,
                              // 20 dwords/row -> 16 lanes hit 16 distinct banks)
#define ROWB  (XR * 2)        // 80 bytes
#define XROWS 128
#define WROWS 192             // Q|K|V weight rows fused (3 * 64)
#define XBUFB (XROWS * ROWB)  // 10240 B
#define WBUFB (WROWS * ROWB)  // 15360 B
#define BUFB  (XBUFB + WBUFB) // 25600 B per buffer, x2 = 50 KB LDS

__device__ __forceinline__ bf16 to_bf16(float f) { return (bf16)f; }

// ---- global fragment loader (ISA 7.12.2 layout) ---------------------------
// Element (i,k) of a 16x32 bf16 tile lives at p[i*rs + k] (row-major, cs==1).
__device__ __forceinline__ v16bf load_frag(const bf16* __restrict__ p, int rs, int lane) {
  const int i  = lane & 15;
  const int h8 = (lane >> 4) << 3;
  const bf16* row = p + i * rs + h8;
  v16bf f;
#pragma unroll
  for (int r = 0; r < 8; ++r) {
    const int kb = (r < 4) ? (2 * r) : (8 + 2 * r);   // r>=4: 16 + 2*(r-4)
    f[2 * r]     = row[kb];
    f[2 * r + 1] = row[kb + 1];
  }
  return f;
}

// Fragment gather straight from an LDS byte address (addrspace(3) loads ->
// ds_load_b64 pairs). dwords {0..3, 8..11} relative to the row pointer.
__device__ __forceinline__ v16bf frag_lds(unsigned byteAddr) {
  ldsCU* p = (ldsCU*)(unsigned long long)byteAddr;
  v8u u;
#pragma unroll
  for (int r = 0; r < 8; ++r)
    u[r] = p[(r < 4) ? r : (4 + r)];
  return __builtin_bit_cast(v16bf, u);
}

// ---- CDNA5 async global->LDS (ASYNCcnt) -----------------------------------
__device__ __forceinline__ void async_b128(unsigned lds_off, const bf16* g) {
  asm volatile("global_load_async_to_lds_b128 %0, %1, off"
               :: "v"(lds_off), "v"((unsigned long long)(size_t)g)
               : "memory");
}

// ---- CDNA5 TDM: one 2D-tile DMA to LDS (TENSORcnt) ------------------------
// rows x 32 bf16 tile, row stride Cn elements, LDS rows padded to 80B via
// pad_interval=3 (every 16 dwords) / pad_amount=3 (4 dwords).
__device__ __forceinline__ void tdm_load_tile(unsigned lds_addr, const bf16* g,
                                              int rows) {
  const unsigned long long ga = (unsigned long long)(size_t)g;
  v4u g0;
  g0[0] = 1u;                                        // count=1, user mode
  g0[1] = lds_addr;                                  // D#.lds_addr
  g0[2] = (unsigned)ga;                              // global_addr[31:0]
  g0[3] = (unsigned)((ga >> 32) & 0x01FFFFFFu) | 0x80000000u;  // [56:32] | type=2
  v8i g1;
  g1[0] = (int)((1u << 16) |                         // data_size = 2B
                (1u << 20) |                         // pad_enable
                (3u << 22) |                         // pad_interval: 16 dwords
                (3u << 25));                         // pad_amount: 4 dwords
  g1[1] = (int)((Cn & 0xFFFFu) << 16);               // tensor_dim0[15:0]
  g1[2] = (int)(((unsigned)Cn >> 16) | ((unsigned)(rows & 0xFFFF) << 16)); // dim0 hi | tensor_dim1 lo
  g1[3] = (int)(32u << 16);                          // tensor_dim1 hi=0 | tile_dim0=32
  g1[4] = rows;                                      // tile_dim1 | tile_dim2=0
  g1[5] = Cn;                                        // tensor_dim0_stride[31:0]
  g1[6] = 0;                                         // stride0 hi | stride1 lo
  g1[7] = 0;                                         // stride1 hi
  v4i z4 = {0, 0, 0, 0};
  v8i z8 = {0, 0, 0, 0, 0, 0, 0, 0};
  __builtin_amdgcn_tensor_load_to_lds(g0, g1, z4, z4, z8, 0);
}

// ---------------------------------------------------------------------- cvt
__global__ void cvt_x_kernel(const float* __restrict__ X, bf16* __restrict__ Xb) {
  long i = (long)blockIdx.x * blockDim.x + threadIdx.x;
  Xb[i] = to_bf16(X[i]);
}

// Fused transposed weights: WT[(w*64+h)][c] = W_w[c][h]   (192 x C, bf16)
__global__ void cvt_w_kernel(const float* __restrict__ Wq, const float* __restrict__ Wk,
                             const float* __restrict__ Wv, bf16* __restrict__ WT) {
  int i = blockIdx.x * blockDim.x + threadIdx.x;     // 3*C*H, h fastest
  int h = i % Hn;
  int c = (i / Hn) % Cn;
  int w = i / (Hn * Cn);
  const float* W = (w == 0) ? Wq : (w == 1 ? Wk : Wv);
  WT[((long)w * Hn + h) * Cn + c] = to_bf16(W[c * Hn + h]);
}

// ----------------------------------------------------------------- QKV GEMM
// Block = (b, 128-row slab). K-chunks of 32 staged in LDS, double-buffered:
// X slab by TDM (wave 0, TENSORcnt), W slab by per-lane async (ASYNCcnt).
// Wave w: mtg=w&1 -> 4 M-tiles, wng=w>>1 -> 3 N-tiles; 12 WMMAs per chunk.
__global__ void __launch_bounds__(256)
qkv_wmma(const bf16* __restrict__ X, const bf16* __restrict__ WT,
         const float* __restrict__ bq, const float* __restrict__ bk,
         const float* __restrict__ bv,
         bf16* __restrict__ Q, bf16* __restrict__ K, bf16* __restrict__ VT) {
  __shared__ __align__(16) unsigned smemU[2 * BUFB / 4];
  // Anchor: make the LDS allocation escape so the async/TDM writers (which
  // only see integer offsets) are potential modifiers; keeps 50KB allocated.
  asm volatile("" :: "s"(&smemU[0]) : "memory");

  const int tid  = threadIdx.x;
  const int lane = tid & 31;
  const int wave = tid >> 5;
  const int m128 = blockIdx.x & 15;                  // 128-row slab
  const int b    = blockIdx.x >> 4;

  const bf16* Xrow0 = X + ((long)b * Tn + m128 * 128) * Cn;

  const int mtg = wave & 1;                          // 4 M-tiles: mtg*4 + mi
  const int wng = wave >> 1;                         // 3 N-tiles: wng*3 + wi

  // W chunk: 192 rows x 32 bf16 -> 768 B128 lane-loads, 3 per thread
  auto issueW = [&](int c) {
    const int k0 = c * 32;
    const unsigned base = (unsigned)((c & 1) * BUFB) + XBUFB;
#pragma unroll
    for (int j = 0; j < 3; ++j) {
      const int li = tid + j * 256;                  // 0..767
      const int row = li >> 2, seg = li & 3;
      async_b128(base + row * ROWB + seg * 16,
                 WT + (long)row * Cn + k0 + seg * 8);
    }
  };
  // X chunk: one TDM descriptor (issued by wave 0 only)
  auto issueX = [&](int c) {
    tdm_load_tile((unsigned)((c & 1) * BUFB), Xrow0 + c * 32, XROWS);
  };

  const unsigned fragLane = (unsigned)((lane & 15) * ROWB + ((lane >> 4) << 4));

  v8f acc[4][3] = {};

  if (wave == 0) issueX(0);
  issueW(0);
  const int NC = Cn / 32;                            // 32 chunks
  for (int c = 0; c < NC; ++c) {
    if (c + 1 < NC) {
      if (wave == 0) issueX(c + 1);
      issueW(c + 1);
      if (wave == 0) __builtin_amdgcn_s_wait_tensorcnt((short)1);
      asm volatile("s_wait_asynccnt %0" :: "i"(3) : "memory");
    } else {
      if (wave == 0) __builtin_amdgcn_s_wait_tensorcnt((short)0);
      asm volatile("s_wait_asynccnt %0" :: "i"(0) : "memory");
    }
    __syncthreads();

    const unsigned bufB = (unsigned)((c & 1) * BUFB);

    v16bf a[4], bb[3];
#pragma unroll
    for (int mi = 0; mi < 4; ++mi)
      a[mi] = frag_lds(bufB + (unsigned)((mtg * 4 + mi) * 16) * ROWB + fragLane);
#pragma unroll
    for (int wi = 0; wi < 3; ++wi)
      bb[wi] = frag_lds(bufB + XBUFB + (unsigned)((wng * 3 + wi) * 16) * ROWB + fragLane);
#pragma unroll
    for (int mi = 0; mi < 4; ++mi)
#pragma unroll
      for (int wi = 0; wi < 3; ++wi)
        acc[mi][wi] = __builtin_amdgcn_wmma_f32_16x16x32_bf16(
            false, a[mi], false, bb[wi], (short)0, acc[mi][wi], false, false);

    __syncthreads();                                 // buffer reuse fence
  }

  const int n  = lane & 15;
  const int mh = (lane >> 4) << 3;
#pragma unroll
  for (int wi = 0; wi < 3; ++wi) {
    const int wn    = wng * 3 + wi;                  // 0..11
    const int which = wn >> 2;
    const int nt    = wn & 3;
    const int col   = nt * 16 + n;
    const float* bias = (which == 0) ? bq : (which == 1 ? bk : bv);
    const float bval  = bias[col];
#pragma unroll
    for (int mi = 0; mi < 4; ++mi) {
#pragma unroll
      for (int r = 0; r < 8; ++r) {
        const int row = m128 * 128 + (mtg * 4 + mi) * 16 + mh + r;
        const float v = acc[mi][wi][r] + bval;
        if (which == 2)
          VT[((long)b * Hn + col) * Tn + row] = to_bf16(v);   // V transposed
        else if (which == 0)
          Q[((long)b * Tn + row) * Hn + col] = to_bf16(v);
        else
          K[((long)b * Tn + row) * Hn + col] = to_bf16(v);
      }
    }
  }
}

// --------------------------------------------------------------- Q @ K^T
__global__ void scores_wmma(const bf16* __restrict__ Q, const bf16* __restrict__ K,
                            float* __restrict__ S) {
  const int lane = threadIdx.x & 31;
  const int wave = threadIdx.x >> 5;
  const int mtp  = blockIdx.x & 63;
  const int b    = blockIdx.x >> 6;
  const int mt0  = 2 * mtp, mt1 = 2 * mtp + 1;

  const bf16* Qp = Q + ((long)b * Tn + mt0 * 16) * Hn;
  const bf16* Kb = K + (long)b * Tn * Hn;
  float* Sp = S + (long)b * Tn * Tn;

  const v16bf a00 = load_frag(Qp, Hn, lane);
  const v16bf a01 = load_frag(Qp + 32, Hn, lane);
  const v16bf a10 = load_frag(Qp + 16 * Hn, Hn, lane);
  const v16bf a11 = load_frag(Qp + 16 * Hn + 32, Hn, lane);

  const int n  = lane & 15;
  const int mh = (lane >> 4) << 3;

  for (int nt = wave; nt <= mt1; nt += 8) {
    const bf16* Kp = Kb + (long)(nt * 16) * Hn;
    const v16bf b0 = load_frag(Kp, Hn, lane);
    const v16bf b1 = load_frag(Kp + 32, Hn, lane);
    v8f acc0 = {}, acc1 = {};
    acc0 = __builtin_amdgcn_wmma_f32_16x16x32_bf16(false, a00, false, b0, (short)0, acc0, false, false);
    acc0 = __builtin_amdgcn_wmma_f32_16x16x32_bf16(false, a01, false, b1, (short)0, acc0, false, false);
    acc1 = __builtin_amdgcn_wmma_f32_16x16x32_bf16(false, a10, false, b0, (short)0, acc1, false, false);
    acc1 = __builtin_amdgcn_wmma_f32_16x16x32_bf16(false, a11, false, b1, (short)0, acc1, false, false);
    const int s = nt * 16 + n;
#pragma unroll
    for (int r = 0; r < 8; ++r) {
      Sp[(long)(mt0 * 16 + mh + r) * Tn + s] = acc0[r];
      Sp[(long)(mt1 * 16 + mh + r) * Tn + s] = acc1[r];
    }
  }
}

// --------------------------------------------- column softmax stats (axis=1)
__global__ void col_chunk_stats(const float* __restrict__ S,
                                float* __restrict__ Mc, float* __restrict__ Lc) {
  long idx = (long)blockIdx.x * blockDim.x + threadIdx.x;  // (b, c, s), s fastest
  const int s = idx % Tn;
  const int c = (idx / Tn) % NCHUNK;
  const int b = idx / ((long)Tn * NCHUNK);
  const float* Sp = S + (long)b * Tn * Tn;
  const int t0 = c * CHUNK;
  const int t1 = t0 + CHUNK;
  const int ts = (t0 > s) ? t0 : s;
  float m = -__builtin_inff();
  for (int t = ts; t < t1; ++t) m = fmaxf(m, Sp[(long)t * Tn + s]);
  float l = 0.0f;
  for (int t = ts; t < t1; ++t) l += __expf(Sp[(long)t * Tn + s] - m);
  Mc[idx] = m;
  Lc[idx] = l;
}

__global__ void col_combine(const float* __restrict__ Mc, const float* __restrict__ Lc,
                            float* __restrict__ M, float* __restrict__ R) {
  const int idx = blockIdx.x * blockDim.x + threadIdx.x;   // b*Tn + s
  const int s = idx % Tn;
  const int b = idx / Tn;
  float m = -__builtin_inff();
#pragma unroll
  for (int c = 0; c < NCHUNK; ++c)
    m = fmaxf(m, Mc[((long)b * NCHUNK + c) * Tn + s]);
  float l = 0.0f;
#pragma unroll
  for (int c = 0; c < NCHUNK; ++c) {
    const long o = ((long)b * NCHUNK + c) * Tn + s;
    l += Lc[o] * __expf(Mc[o] - m);                        // empty chunk: 0*0
  }
  M[idx] = m;
  R[idx] = 1.0f / l;
}

// P = exp(S - M[s]) * R[s] (bf16); upper triangle stays 0 from the memset.
__global__ void norm_exp(const float* __restrict__ S, const float* __restrict__ M,
                         const float* __restrict__ R, bf16* __restrict__ P) {
  long idx = (long)blockIdx.x * blockDim.x + threadIdx.x;  // ((b*T)+t)*T + s
  const int s = idx % Tn;
  const int t = (idx / Tn) % Tn;
  const int b = idx / ((long)Tn * Tn);
  if (t < s) return;
  const int cs = b * Tn + s;
  P[idx] = to_bf16(__expf(S[idx] - M[cs]) * R[cs]);
}

// ------------------------------------------------------------------ P @ V
__global__ void out_wmma(const bf16* __restrict__ P, const bf16* __restrict__ VT,
                         float* __restrict__ O) {
  const int lane = threadIdx.x & 31;
  const int gw   = blockIdx.x * (blockDim.x >> 5) + (threadIdx.x >> 5);
  const int nt = gw & 3;
  const int q  = (gw >> 2) & 31;                     // 4 M-tiles: 4q..4q+3
  const int b  = gw >> 7;

  const bf16* Pb = P  + ((long)b * Tn + q * 64) * Tn;
  const bf16* Vp = VT + ((long)b * Hn + nt * 16) * Tn;

  v8f acc[4] = {};
  const int kmax = 2 * q + 1;                        // covers s <= 64q+63
  for (int kt = 0; kt <= kmax; ++kt) {
    const int k0 = kt * 32;
    __builtin_prefetch(Vp + k0 + 64, 0, 1);
    const v16bf bb = load_frag(Vp + k0, Tn, lane);
#pragma unroll
    for (int mi = 0; mi < 4; ++mi) {
      const v16bf a = load_frag(Pb + (long)(mi * 16) * Tn + k0, Tn, lane);
      acc[mi] = __builtin_amdgcn_wmma_f32_16x16x32_bf16(false, a, false, bb,
                                                        (short)0, acc[mi], false, false);
    }
  }
  const int n  = lane & 15;
  const int mh = (lane >> 4) << 3;
#pragma unroll
  for (int mi = 0; mi < 4; ++mi) {
#pragma unroll
    for (int r = 0; r < 8; ++r) {
      const int t = q * 64 + mi * 16 + mh + r;
      O[((long)b * Tn + t) * Hn + nt * 16 + n] = acc[mi][r];
    }
  }
}

// ---------------------------------------------------------------------------
extern "C" void kernel_launch(void* const* d_in, const int* in_sizes, int n_in,
                              void* d_out, int out_size, void* d_ws, size_t ws_size,
                              hipStream_t stream) {
  const float* x  = (const float*)d_in[0];
  const float* Wq = (const float*)d_in[1];
  const float* bq = (const float*)d_in[2];
  const float* Wk = (const float*)d_in[3];
  const float* bk = (const float*)d_in[4];
  const float* Wv = (const float*)d_in[5];
  const float* bv = (const float*)d_in[6];
  float* out = (float*)d_out;

  char* ws = (char*)d_ws;
  size_t off = 0;
  auto alloc = [&](size_t bytes) -> char* {
    char* p = ws + off;
    off += (bytes + 255) & ~(size_t)255;
    return p;
  };

  bf16*  Xb   = (bf16*) alloc((size_t)Bn * Tn * Cn * 2);       //  67 MB
  bf16*  WT   = (bf16*) alloc((size_t)3 * Hn * Cn * 2);        // 384 KB
  bf16*  Qb   = (bf16*) alloc((size_t)Bn * Tn * Hn * 2);       //   4 MB
  bf16*  Kb   = (bf16*) alloc((size_t)Bn * Tn * Hn * 2);       //   4 MB
  bf16*  VT   = (bf16*) alloc((size_t)Bn * Hn * Tn * 2);       //   4 MB
  float* S    = (float*)alloc((size_t)Bn * Tn * Tn * 4);       // 268 MB
  bf16*  P    = (bf16*) alloc((size_t)Bn * Tn * Tn * 2);       // 134 MB
  float* Mc   = (float*)alloc((size_t)Bn * Tn * NCHUNK * 4);   //   1 MB
  float* Lc   = (float*)alloc((size_t)Bn * Tn * NCHUNK * 4);   //   1 MB
  float* Mcol = (float*)alloc((size_t)Bn * Tn * 4);
  float* Rcol = (float*)alloc((size_t)Bn * Tn * 4);

  // Upper-triangle probabilities must be exactly zero (ws is poisoned).
  (void)hipMemsetAsync(P, 0, (size_t)Bn * Tn * Tn * 2, stream);

  cvt_x_kernel<<<(Bn * Tn * Cn) / 256, 256, 0, stream>>>(x, Xb);
  cvt_w_kernel<<<(3 * Cn * Hn) / 256, 256, 0, stream>>>(Wq, Wk, Wv, WT);

  qkv_wmma<<<Bn * (Tn / 128), 256, 0, stream>>>(Xb, WT, bq, bk, bv, Qb, Kb, VT);

  scores_wmma<<<Bn * (Tn / 32), 256, 0, stream>>>(Qb, Kb, S);

  col_chunk_stats<<<(Bn * Tn * NCHUNK) / 256, 256, 0, stream>>>(S, Mc, Lc);
  col_combine<<<(Bn * Tn) / 256, 256, 0, stream>>>(Mc, Lc, Mcol, Rcol);

  norm_exp<<<(unsigned)(((size_t)Bn * Tn * Tn) / 256), 256, 0, stream>>>(
      S, Mcol, Rcol, P);

  out_wmma<<<(Bn * (Tn / 16) * (Hn / 16)) / 8 / 4, 256, 0, stream>>>(P, VT, out);
}
